// TransformerBlock_61546881351760
// MI455X (gfx1250) — matrix operand order repaired and checked by reading, not compiled, not run on hardware
//
#include <hip/hip_runtime.h>
#include <hip/hip_bf16.h>
#include <math.h>

// ---------------------------------------------------------------- constants
constexpr int Bc  = 2;
constexpr int Tc  = 2048;
constexpr int Cc  = 1024;
constexpr int Hc  = 16;
constexpr int HDc = 64;          // head dim
constexpr int Mtot = Bc * Tc;    // 4096 rows
constexpr int C4  = 4 * Cc;      // 4096
constexpr float EPSc = 1e-5f;

typedef __attribute__((ext_vector_type(16))) __bf16 v16bf;
typedef __attribute__((ext_vector_type(8)))  float  v8f;

// ---------------------------------------------------------------- helpers
__device__ __forceinline__ __bf16 to_bf16(float f) {
    unsigned u = __builtin_bit_cast(unsigned, f);
    unsigned r = (u + 0x7fffu + ((u >> 16) & 1u)) >> 16;   // RNE
    unsigned short us = (unsigned short)r;
    return __builtin_bit_cast(__bf16, us);
}

// Async copy: 16 bytes global -> LDS, tracked by ASYNCcnt (CDNA5 VFLAT async).
__device__ __forceinline__ void async_ld_b128(unsigned ldsOff, const void* gptr) {
    asm volatile("global_load_async_to_lds_b128 %0, %1, off"
                 :: "v"(ldsOff), "v"(gptr) : "memory");
}
__device__ __forceinline__ void wait_async0() {
    asm volatile("s_wait_asynccnt 0x0" ::: "memory");
}
__device__ __forceinline__ unsigned lds_addr(const void* p) {
    return (unsigned)(uintptr_t)p;   // LDS aperture: addr[31:0] = LDS offset
}

// A-fragment (16-bit A 16x32 layout): lane holds row m=lane&15;
// K elems are 16 contiguous at kb, kb = (lane<16)?0:16.
__device__ __forceinline__ v16bf ldfragA(const __bf16* p, int kb) {
    union { v16bf v; uint4 q[2]; } u;
    u.q[0] = *(const uint4*)(p + kb);
    u.q[1] = *(const uint4*)(p + kb + 8);
    return u.v;
}
// B-fragment from [n][k]-major LDS: lane holds col n=lane&15;
// K elems are 16 contiguous at kb2 = (lane<16)?0:16.
__device__ __forceinline__ v16bf ldfragB(const __bf16* p) {
    union { v16bf v; uint4 q[2]; } u;
    u.q[0] = *(const uint4*)(p);
    u.q[1] = *(const uint4*)(p + 8);
    return u.v;
}

__device__ __forceinline__ v8f wmma_bf16(v16bf a, v16bf b, v8f c) {
    return __builtin_amdgcn_wmma_f32_16x16x32_bf16(false, a, false, b, (short)0, c,
                                                   false, false);
}

// ---------------------------------------------------------------- f32 -> bf16
__global__ __launch_bounds__(256) void cvt_kernel(const float* __restrict__ in,
                                                  __bf16* __restrict__ out, int n) {
    int i = blockIdx.x * 256 + threadIdx.x;
    if (i < n) out[i] = to_bf16(in[i]);
}

// ---------------------------------------------------------------- generic GEMM
// Block tile 256x64, K-step 32, 256 threads (8 waves); wave = 32x64 strip
// (2 A-frags x 4 B-frags -> 8 WMMAs per K-step, B-frags reused).
// Double-buffered LDS; A tile staged with async-to-LDS (4x b128/thread),
// B tile staged transposed via VGPRs with the global load issued one
// iteration ahead.
constexpr int BM = 256, BN = 64, BK = 32;
constexpr int LAS = 40;   // LDS row stride (elems) for A  (80B, 16B multiple)
constexpr int LBS = 40;   // LDS row stride for B^T

__global__ __launch_bounds__(256)
void gemm_bf16_kernel(const __bf16* __restrict__ A, const __bf16* __restrict__ Bm,
                      float* __restrict__ outF, __bf16* __restrict__ outB,
                      const float* __restrict__ bias, const float* __restrict__ resid,
                      int lda, int ldb, int ldc, int K, int relu,
                      int batchMod, size_t aOuter, size_t aInner,
                      size_t bInner, size_t cStride) {
    __shared__ __align__(16) __bf16 As[2][BM * LAS];
    __shared__ __align__(16) __bf16 Bs[2][BN * LBS];

    const int tid  = threadIdx.x;
    const int wv   = tid >> 5;
    const int lane = tid & 31;
    const int m    = lane & 15;
    const int hlf  = lane >> 4;
    const int kbA  = hlf * 16;
    const int kbB  = hlf * 16;

    {   // batch indexing (QKV: z = b*H + h)
        int z  = blockIdx.z;
        int zo = z / batchMod, zi = z - zo * batchMod;
        A  += zo * aOuter + zi * aInner;
        Bm += (size_t)zi * bInner;
        size_t co = (size_t)z * cStride;
        if (outF)  outF  += co;
        if (outB)  outB  += co;
        if (resid) resid += co;
    }

    const int rowBase = blockIdx.x * BM;
    const int colBase = blockIdx.y * BN;

    // staging coordinates: A -> thread owns one full 32-elem K-row (4x b128)
    const int rA  = tid;                               // row 0..255
    const int kkB = tid >> 3, n0B = (tid & 7) * 8;     // B: 8 elems/thread

    v8f acc[2][4];
    #pragma unroll
    for (int s = 0; s < 2; ++s)
        #pragma unroll
        for (int j = 0; j < 4; ++j) acc[s][j] = (v8f)(0.f);

    auto stageA = [&](int k0, int buf) {
        const __bf16* g = A + (size_t)(rowBase + rA) * lda + k0;
        unsigned l = lds_addr(&As[buf][rA * LAS]);
        async_ld_b128(l,      g);
        async_ld_b128(l + 16, g + 8);
        async_ld_b128(l + 32, g + 16);
        async_ld_b128(l + 48, g + 24);
    };
    auto stageB = [&](uint4 d, int buf) {
        const __bf16* pv = (const __bf16*)&d;
        #pragma unroll
        for (int i = 0; i < 8; ++i) Bs[buf][(n0B + i) * LBS + kkB] = pv[i];
    };

    // prologue: stage tile 0 into buffer 0
    stageA(0, 0);
    stageB(*(const uint4*)(Bm + (size_t)kkB * ldb + colBase + n0B), 0);

    const int nIter = K / BK;
    for (int it = 0; it < nIter; ++it) {
        const int cb = it & 1;
        const bool hasNext = (it + 1) < nIter;
        uint4 bnext = {0, 0, 0, 0};
        if (hasNext)   // start next B global load early (hides HBM latency)
            bnext = *(const uint4*)(Bm + (size_t)((it + 1) * BK + kkB) * ldb +
                                    colBase + n0B);

        wait_async0();        // my async A copies for tile `it` have landed
        __syncthreads();      // everyone's copies/stores for tile `it` visible

        v16bf af[2];
        #pragma unroll
        for (int s = 0; s < 2; ++s)
            af[s] = ldfragA(&As[cb][(wv * 32 + s * 16 + m) * LAS], kbA);
        v16bf bfr[4];
        #pragma unroll
        for (int j = 0; j < 4; ++j)
            bfr[j] = ldfragB(&Bs[cb][(j * 16 + m) * LBS + kbB]);
        #pragma unroll
        for (int s = 0; s < 2; ++s)
            #pragma unroll
            for (int j = 0; j < 4; ++j)
                acc[s][j] = wmma_bf16(af[s], bfr[j], acc[s][j]);

        __syncthreads();      // all waves done reading the buffer to be rewritten

        if (hasNext) {
            stageA((it + 1) * BK, cb ^ 1);
            stageB(bnext, cb ^ 1);
        }
    }

    // epilogue: C layout -> lane col n = lane&15, rows hlf*8 + r
    #pragma unroll
    for (int s = 0; s < 2; ++s)
        #pragma unroll
        for (int j = 0; j < 4; ++j) {
            int n = colBase + j * 16 + m;
            float bv = bias ? bias[n] : 0.f;
            #pragma unroll
            for (int r = 0; r < 8; ++r) {
                int row = rowBase + wv * 32 + s * 16 + hlf * 8 + r;
                float vvl = acc[s][j][r] + bv;
                if (resid) vvl += resid[(size_t)row * ldc + n];
                if (relu)  vvl = vvl > 0.f ? vvl : 0.f;
                if (outF)  outF[(size_t)row * ldc + n] = vvl;
                if (outB)  outB[(size_t)row * ldc + n] = to_bf16(vvl);
            }
        }
}

// ---------------------------------------------------------------- flash attention
// grid: (T/128, B*H); 8 waves, each wave = 16 query rows; streams 32-key tiles.
constexpr int KSS = 72;   // Ks row stride (144B)
constexpr int VSS = 40;   // Vs / Ps row stride (80B)

__global__ __launch_bounds__(256)
void attn_kernel(const __bf16* __restrict__ q, const __bf16* __restrict__ k,
                 const __bf16* __restrict__ v, __bf16* __restrict__ obf) {
    __shared__ __align__(16) __bf16 Ks[32 * KSS];          // [key][d]  == B-layout for Q@K^T
    __shared__ __align__(16) __bf16 Vs[64 * VSS];          // [d][key]  == B-layout for P@V
    __shared__ __align__(16) __bf16 Ps[8 * 16 * VSS];      // per-wave P tile [16][32]

    const int tid  = threadIdx.x;
    const int wv   = tid >> 5;
    const int lane = tid & 31;
    const int m    = lane & 15;
    const int hlf  = lane >> 4;
    const int kbA  = hlf * 16;
    const int kbB  = hlf * 16;

    const int bh = blockIdx.y;
    const int bb = bh / Hc, hh = bh - bb * Hc;
    const size_t base = (size_t)bh * Tc * HDc;
    q += base; k += base; v += base;

    const int qtile = blockIdx.x * 128;
    const int qrow  = qtile + wv * 16;

    // Q fragments (contraction d = 64 -> two 32-K frags)
    const __bf16* qp = q + (size_t)(qrow + m) * HDc;
    v16bf q0 = ldfragA(qp, kbA);
    v16bf q1 = ldfragA(qp + 32, kbA);

    v8f o[4];
    #pragma unroll
    for (int j = 0; j < 4; ++j) o[j] = (v8f)(0.f);
    float mrow[8], lrow[8];
    #pragma unroll
    for (int r = 0; r < 8; ++r) { mrow[r] = -__builtin_inff(); lrow[r] = 0.f; }

    const int keyT = tid >> 3, d0T = (tid & 7) * 8;   // staging coords

    const int nkt = (blockIdx.x + 1) * 4;   // 32-key tiles, causal bound (uniform)
    for (int kt = 0; kt < nkt; ++kt) {
        const int s0 = kt * 32;
        __syncthreads();
        // K tile [32][64] row-major (already B-layout [n=key][k=d]): async copy
        async_ld_b128(lds_addr(&Ks[keyT * KSS + d0T]),
                      k + (size_t)(s0 + keyT) * HDc + d0T);
        {   // V tile transposed -> Vs[d][key] (VGPR path, 16-bit scatter)
            const __bf16* g = v + (size_t)(s0 + keyT) * HDc + d0T;
            uint4 vv = *(const uint4*)g;
            const __bf16* pv = (const __bf16*)&vv;
            #pragma unroll
            for (int i = 0; i < 8; ++i) Vs[(d0T + i) * VSS + keyT] = pv[i];
        }
        wait_async0();
        __syncthreads();

        // scores: two 16x16 tiles over this 32-key chunk
        v8f s[2];
        #pragma unroll
        for (int j = 0; j < 2; ++j) {
            const __bf16* kp = &Ks[(j * 16 + m) * KSS];
            v16bf kf0 = ldfragB(kp + kbB);
            v16bf kf1 = ldfragB(kp + 32 + kbB);
            v8f sa = (v8f)(0.f);
            sa = wmma_bf16(q0, kf0, sa);
            sa = wmma_bf16(q1, kf1, sa);
            int sg = s0 + j * 16 + m;
            #pragma unroll
            for (int r = 0; r < 8; ++r) {
                int tg = qrow + hlf * 8 + r;
                float sv = sa[r] * 0.125f;                 // 1/sqrt(64)
                s[j][r] = (sg > tg) ? -__builtin_inff() : sv;
            }
        }

        // online softmax (row stats via 16-lane-half shfl reductions)
        #pragma unroll
        for (int r = 0; r < 8; ++r) {
            float mx = fmaxf(s[0][r], s[1][r]);
            mx = fmaxf(mx, __shfl_xor(mx, 1, 32));
            mx = fmaxf(mx, __shfl_xor(mx, 2, 32));
            mx = fmaxf(mx, __shfl_xor(mx, 4, 32));
            mx = fmaxf(mx, __shfl_xor(mx, 8, 32));
            float nm = fmaxf(mrow[r], mx);
            float al = __expf(mrow[r] - nm);
            float p0 = __expf(s[0][r] - nm);
            float p1 = __expf(s[1][r] - nm);
            s[0][r] = p0; s[1][r] = p1;
            float sm = p0 + p1;
            sm += __shfl_xor(sm, 1, 32);
            sm += __shfl_xor(sm, 2, 32);
            sm += __shfl_xor(sm, 4, 32);
            sm += __shfl_xor(sm, 8, 32);
            lrow[r] = lrow[r] * al + sm;
            mrow[r] = nm;
            #pragma unroll
            for (int j = 0; j < 4; ++j) o[j][r] *= al;
        }

        // P (C layout) -> LDS bf16 (A layout source), then o += P @ V
        __bf16* pw = &Ps[wv * 16 * VSS];
        #pragma unroll
        for (int j = 0; j < 2; ++j)
            #pragma unroll
            for (int r = 0; r < 8; ++r)
                pw[(hlf * 8 + r) * VSS + j * 16 + m] = to_bf16(s[j][r]);
        __syncthreads();
        v16bf pf = ldfragA(&pw[m * VSS], kbA);
        v16bf vfr[4];
        #pragma unroll
        for (int j = 0; j < 4; ++j)
            vfr[j] = ldfragB(&Vs[(j * 16 + m) * VSS + kbB]);
        #pragma unroll
        for (int j = 0; j < 4; ++j)
            o[j] = wmma_bf16(pf, vfr[j], o[j]);
    }

    // normalize + write head-concatenated [B*T, C] bf16
    #pragma unroll
    for (int j = 0; j < 4; ++j)
        #pragma unroll
        for (int r = 0; r < 8; ++r) {
            int trow = qrow + hlf * 8 + r;
            float val = o[j][r] / lrow[r];
            obf[(size_t)(bb * Tc + trow) * Cc + hh * HDc + j * 16 + m] = to_bf16(val);
        }
}

// ---------------------------------------------------------------- layernorm
__global__ __launch_bounds__(256)
void ln_kernel(const float* __restrict__ in, const float* __restrict__ gamma,
               const float* __restrict__ beta, float* __restrict__ outF,
               __bf16* __restrict__ outB) {
    __shared__ float red[256];
    const int row = blockIdx.x, tid = threadIdx.x;
    const float* x = in + (size_t)row * Cc;
    float lx[4];
    float s = 0.f;
    #pragma unroll
    for (int i = 0; i < 4; ++i) { lx[i] = x[tid + i * 256]; s += lx[i]; }
    red[tid] = s; __syncthreads();
    for (int off = 128; off > 0; off >>= 1) {
        if (tid < off) red[tid] += red[tid + off];
        __syncthreads();
    }
    float mu = red[0] * (1.f / Cc);
    __syncthreads();
    float vs = 0.f;
    #pragma unroll
    for (int i = 0; i < 4; ++i) { float d = lx[i] - mu; vs += d * d; }
    red[tid] = vs; __syncthreads();
    for (int off = 128; off > 0; off >>= 1) {
        if (tid < off) red[tid] += red[tid + off];
        __syncthreads();
    }
    float inv = rsqrtf(red[0] * (1.f / Cc) + EPSc);
    #pragma unroll
    for (int i = 0; i < 4; ++i) {
        int c = tid + i * 256;
        float y = (lx[i] - mu) * inv * gamma[c] + beta[c];
        if (outF) outF[(size_t)row * Cc + c] = y;
        if (outB) outB[(size_t)row * Cc + c] = to_bf16(y);
    }
}

// ---------------------------------------------------------------- launch
extern "C" void kernel_launch(void* const* d_in, const int* in_sizes, int n_in,
                              void* d_out, int out_size, void* d_ws, size_t ws_size,
                              hipStream_t stream) {
    const float* x   = (const float*)d_in[0];
    const float* Wq  = (const float*)d_in[1];
    const float* Wk  = (const float*)d_in[2];
    const float* Wv  = (const float*)d_in[3];
    const float* Wp  = (const float*)d_in[4];
    const float* bp  = (const float*)d_in[5];
    const float* W1  = (const float*)d_in[6];
    const float* b1  = (const float*)d_in[7];
    const float* W2  = (const float*)d_in[8];
    const float* b2  = (const float*)d_in[9];
    const float* g1  = (const float*)d_in[10];
    const float* be1 = (const float*)d_in[11];
    const float* g2  = (const float*)d_in[12];
    const float* be2 = (const float*)d_in[13];

    char* ws = (char*)d_ws;
    size_t off = 0;
    auto alloc = [&](size_t bytes) {
        size_t o = off; off += (bytes + 255) & ~(size_t)255; return o;
    };
    const size_t nX  = (size_t)Mtot * Cc;          // 4M elems
    const size_t nWh = (size_t)Hc * Cc * HDc;      // 1M per head-stack
    const size_t nQ  = (size_t)Bc * Hc * Tc * HDc; // 4M

    size_t oXbf = alloc(nX * 2);                   // reused as obf after QKV
    size_t oWqb = alloc(nWh * 2);
    size_t oWkb = alloc(nWh * 2);
    size_t oWvb = alloc(nWh * 2);
    size_t oWpb = alloc((size_t)Cc * Cc * 2);
    size_t oW1b = alloc((size_t)Cc * C4 * 2);
    size_t oW2b = alloc((size_t)C4 * Cc * 2);
    size_t oQ   = alloc(nQ * 2);                   // oQ..oQ+32MB reused as h1
    size_t oK   = alloc(nQ * 2);
    size_t oV   = alloc(nQ * 2);
    (void)       alloc(nQ * 2);                    // pad so h1 (32MB) fits at oQ
    size_t oRes = alloc((size_t)Mtot * Cc * 4);    // res1, reused as res2
    size_t oLnF = alloc((size_t)Mtot * Cc * 4);
    size_t oLnB = alloc((size_t)Mtot * Cc * 2);

    __bf16* Xbf = (__bf16*)(ws + oXbf);
    __bf16* Wqb = (__bf16*)(ws + oWqb);
    __bf16* Wkb = (__bf16*)(ws + oWkb);
    __bf16* Wvb = (__bf16*)(ws + oWvb);
    __bf16* Wpb = (__bf16*)(ws + oWpb);
    __bf16* W1b = (__bf16*)(ws + oW1b);
    __bf16* W2b = (__bf16*)(ws + oW2b);
    __bf16* qb  = (__bf16*)(ws + oQ);
    __bf16* kb  = (__bf16*)(ws + oK);
    __bf16* vb  = (__bf16*)(ws + oV);
    __bf16* obf = Xbf;                             // alias (Xbf dead after QKV)
    __bf16* h1b = qb;                              // alias (q/k/v dead after attn)
    float*  res = (float*)(ws + oRes);
    float*  lnF = (float*)(ws + oLnF);
    __bf16* lnB = (__bf16*)(ws + oLnB);

    auto cvt = [&](const float* src, __bf16* dst, size_t n) {
        cvt_kernel<<<dim3((unsigned)((n + 255) / 256)), dim3(256), 0, stream>>>(src, dst, (int)n);
    };
    cvt(x,  Xbf, nX);
    cvt(Wq, Wqb, nWh);
    cvt(Wk, Wkb, nWh);
    cvt(Wv, Wvb, nWh);
    cvt(Wp, Wpb, (size_t)Cc * Cc);
    cvt(W1, W1b, (size_t)Cc * C4);
    cvt(W2, W2b, (size_t)C4 * Cc);

    // QKV: per (b,h) GEMM  [T,C] @ [C,HD] -> [T,HD] bf16
    dim3 gQKV(Tc / BM, 1, Bc * Hc);
    const size_t aOut = (size_t)Tc * Cc, bIn = (size_t)Cc * HDc, cStr = (size_t)Tc * HDc;
    gemm_bf16_kernel<<<gQKV, 256, 0, stream>>>(Xbf, Wqb, nullptr, qb, nullptr, nullptr,
        Cc, HDc, HDc, Cc, 0, Hc, aOut, 0, bIn, cStr);
    gemm_bf16_kernel<<<gQKV, 256, 0, stream>>>(Xbf, Wkb, nullptr, kb, nullptr, nullptr,
        Cc, HDc, HDc, Cc, 0, Hc, aOut, 0, bIn, cStr);
    gemm_bf16_kernel<<<gQKV, 256, 0, stream>>>(Xbf, Wvb, nullptr, vb, nullptr, nullptr,
        Cc, HDc, HDc, Cc, 0, Hc, aOut, 0, bIn, cStr);

    // causal flash attention -> obf [B*T, C] bf16 (head-concat)
    attn_kernel<<<dim3(Tc / 128, Bc * Hc), 256, 0, stream>>>(qb, kb, vb, obf);

    // proj: obf @ Wp + bp + x  -> res (f32)
    gemm_bf16_kernel<<<dim3(Mtot / BM, Cc / BN, 1), 256, 0, stream>>>(
        obf, Wpb, res, nullptr, bp, x, Cc, Cc, Cc, Cc, 0, 1, 0, 0, 0, 0);

    // LN1 -> lnF (f32 residual src) + lnB (bf16 FFN input)
    ln_kernel<<<dim3(Mtot), 256, 0, stream>>>(res, g1, be1, lnF, lnB);

    // FFN1: relu(lnB @ W1 + b1) -> h1b bf16
    gemm_bf16_kernel<<<dim3(Mtot / BM, C4 / BN, 1), 256, 0, stream>>>(
        lnB, W1b, nullptr, h1b, b1, nullptr, Cc, C4, C4, Cc, 1, 1, 0, 0, 0, 0);

    // FFN2: h1b @ W2 + b2 + lnF -> res (f32)
    gemm_bf16_kernel<<<dim3(Mtot / BM, Cc / BN, 1), 256, 0, stream>>>(
        h1b, W2b, res, nullptr, b2, lnF, C4, Cc, Cc, C4, 0, 1, 0, 0, 0, 0);

    // LN2 -> d_out (f32)
    ln_kernel<<<dim3(Mtot), 256, 0, stream>>>(res, g2, be2, (float*)d_out, nullptr);
}
